// BoxBottleneck_82712480187053
// MI455X (gfx1250) — matrix operand order repaired and checked by reading, not compiled, not run on hardware
//
#include <hip/hip_runtime.h>

typedef __attribute__((ext_vector_type(16))) _Float16 v16h;
typedef __attribute__((ext_vector_type(8)))  float    v8f;
typedef __attribute__((ext_vector_type(4)))  unsigned int v4u;
typedef __attribute__((ext_vector_type(8)))  int      v8i;
typedef __attribute__((ext_vector_type(4)))  int      v4i;

#define HH   56
#define WW_  56
#define HW   (56*56)
#define IW   57
#define IHW  (57*57)
#define NB   32
#define CIN  256
#define CMID 16
#define CBOX 64
#define COUT 256
#define EPSL 1e-5f
#define W3PAD 68   // 64 dwords + 4 pad dwords per logical row (bank-conflict-free)

// LDS byte offset of a generic pointer known to point into LDS.
__device__ __forceinline__ unsigned lds_offset(const void* p) {
  return (unsigned)(unsigned long long)
         (__attribute__((address_space(3))) const void*)p;
}

// ---------------------------------------------------------------------------
// Kernel A: 1x1 conv (w1: 16x256) + BN1 + ReLU via WMMA f16.
// One wave per 16(Cmid) x 64(pixels): 4 N-tiles share one A fragment per
// K-step -> 4 back-to-back WMMAs, K=256 in 8 steps.  x is streamed with
// non-temporal loads to keep L2 for the reuse-heavy intermediates.
// ---------------------------------------------------------------------------
__global__ void __launch_bounds__(256) k_conv1(
    const float* __restrict__ x, const float* __restrict__ w1,
    const float* __restrict__ g1, const float* __restrict__ bb1,
    const float* __restrict__ m1, const float* __restrict__ v1,
    float* __restrict__ mid) {
  __shared__ float s_sc[CMID], s_sh[CMID];
  int tid = threadIdx.x;
  if (tid < CMID) {
    float inv = g1[tid] * rsqrtf(v1[tid] + EPSL);
    s_sc[tid] = inv;
    s_sh[tid] = bb1[tid] - m1[tid] * inv;
  }
  __syncthreads();

  int wave = (int)(blockIdx.x * blockDim.x + tid) >> 5;
  int lane = tid & 31;
  int l16  = lane & 15;
  int half = lane >> 4;
  int n  = wave / (HW / 64);          // 49 groups of 64 pixels per image
  int p0 = (wave % (HW / 64)) * 64;

  const float* xp = x + (size_t)n * CIN * HW + p0 + l16;
  v8f acc0 = {}, acc1 = {}, acc2 = {}, acc3 = {};
  for (int k0 = 0; k0 < CIN; k0 += 32) {
    v16h a;
    const float* wr = w1 + l16 * CIN + k0 + half * 8;
    #pragma unroll
    for (int i = 0; i < 8; ++i) {
      a[i]     = (_Float16)wr[i];
      a[i + 8] = (_Float16)wr[16 + i];
    }
    v16h b0, b1, b2, b3;
    int kb = k0 + half * 8;
    #pragma unroll
    for (int i = 0; i < 16; ++i) {
      const float* col = xp + (size_t)(kb + i + (i & 8)) * HW;
      b0[i] = (_Float16)__builtin_nontemporal_load(col);
      b1[i] = (_Float16)__builtin_nontemporal_load(col + 16);
      b2[i] = (_Float16)__builtin_nontemporal_load(col + 32);
      b3[i] = (_Float16)__builtin_nontemporal_load(col + 48);
    }
    acc0 = __builtin_amdgcn_wmma_f32_16x16x32_f16(false, a, false, b0, (short)0, acc0, false, false);
    acc1 = __builtin_amdgcn_wmma_f32_16x16x32_f16(false, a, false, b1, (short)0, acc1, false, false);
    acc2 = __builtin_amdgcn_wmma_f32_16x16x32_f16(false, a, false, b2, (short)0, acc2, false, false);
    acc3 = __builtin_amdgcn_wmma_f32_16x16x32_f16(false, a, false, b3, (short)0, acc3, false, false);
  }
  float* mp = mid + (size_t)n * CMID * HW + p0 + l16;
  #pragma unroll
  for (int r = 0; r < 8; ++r) {
    int m = r + half * 8;
    float sc = s_sc[m], sh = s_sh[m];
    float* row = mp + (size_t)m * HW;
    row[0]  = fmaxf(acc0[r] * sc + sh, 0.f);
    row[16] = fmaxf(acc1[r] * sc + sh, 0.f);
    row[32] = fmaxf(acc2[r] * sc + sh, 0.f);
    row[48] = fmaxf(acc3[r] * sc + sh, 0.f);
  }
}

// ---------------------------------------------------------------------------
// Kernel B: padded 57x57 integral image per (n, cmid) plane.
// The source plane is staged into LDS with ASYNC global->LDS copies
// (coalesced DMA path, tracked by ASYNCcnt) before the prefix sums.
// ---------------------------------------------------------------------------
__global__ void __launch_bounds__(64) k_integral(
    const float* __restrict__ mid, float* __restrict__ I) {
  __shared__ float sM[HW];
  __shared__ float sI[IHW];
  int plane = blockIdx.x;          // NB*CMID = 512 planes
  int t = threadIdx.x;             // 64 threads (2 waves)

  const float* src = mid + (size_t)plane * HW;
  for (int i = t; i < HW; i += 64) {
    unsigned lofs = lds_offset(&sM[i]);
    unsigned long long ga = (unsigned long long)(uintptr_t)(src + i);
    asm volatile("global_load_async_to_lds_b32 %0, %1, off"
                 :: "v"(lofs), "v"(ga) : "memory");
  }
  for (int i = t; i < IHW; i += 64) sI[i] = 0.f;
  asm volatile("s_wait_asynccnt 0x0" ::: "memory");
  __syncthreads();

  if (t < HH) {                    // row-wise prefix sums
    float run = 0.f;
    #pragma unroll 4
    for (int xc = 0; xc < WW_; ++xc) {
      run += sM[t * WW_ + xc];
      sI[(t + 1) * IW + xc + 1] = run;
    }
  }
  __syncthreads();
  if (t < IW) {                    // column-wise prefix sums
    float run = 0.f;
    #pragma unroll 4
    for (int y = 0; y < IW; ++y) {
      run += sI[y * IW + t];
      sI[y * IW + t] = run;
    }
  }
  __syncthreads();
  float* dst = I + (size_t)plane * IHW;
  for (int i = t; i < IHW; i += 64) dst[i] = sI[i];
}

// Bilinear corner lookup on the integral image (matches reference clipping).
__device__ __forceinline__ float corner(const float* __restrict__ Ip,
                                        float cy, float cx) {
  int iy0 = (int)floorf(cy); iy0 = iy0 < 0 ? 0 : (iy0 > HH - 1 ? HH - 1 : iy0);
  int ix0 = (int)floorf(cx); ix0 = ix0 < 0 ? 0 : (ix0 > WW_ - 1 ? WW_ - 1 : ix0);
  float ty = cy - (float)iy0;
  float tx = cx - (float)ix0;
  const float* r0 = Ip + iy0 * IW + ix0;
  const float* r1 = r0 + IW;
  float a = r0[0], b = r0[1], c = r1[0], d = r1[1];
  float top = a + tx * (b - a);
  float bot = c + tx * (d - c);
  return top + ty * (bot - top);
}

// ---------------------------------------------------------------------------
// Kernel C: fused box-conv gather + BN2 + ReLU + 1x1 conv (w3: 256x64) via
// WMMA f16 + BN3 + residual + ReLU.  w3 (64 KB) is staged into LDS by the
// Tensor Data Mover in ONE descriptor, using D# pad_enable (4 dwords pad per
// 64) so A-fragment gathers are bank-conflict-free (stride 68 dwords).
// ---------------------------------------------------------------------------
__global__ void __launch_bounds__(256) k_box_gemm(
    const float* __restrict__ I, const float* __restrict__ x,
    const float* __restrict__ ymn_, const float* __restrict__ ymx_,
    const float* __restrict__ xmn_, const float* __restrict__ xmx_,
    const float* __restrict__ g2, const float* __restrict__ bb2,
    const float* __restrict__ m2, const float* __restrict__ v2,
    const float* __restrict__ w3, const float* __restrict__ g3,
    const float* __restrict__ bb3, const float* __restrict__ m3,
    const float* __restrict__ v3, float* __restrict__ out) {
  __shared__ float sW3f[COUT * W3PAD];     // 68 KB padded f32 weight stage
  __shared__ float s_sc2[CBOX], s_sh2[CBOX];
  __shared__ float s_sc3[COUT], s_sh3[COUT];
  __shared__ float s_box[CBOX][4];
  int tid = threadIdx.x;

#if __has_builtin(__builtin_amdgcn_tensor_load_to_lds)
  if (tid < 32) {   // wave 0 issues one TDM descriptor for the whole 64 KB
    unsigned long long ga = (unsigned long long)(uintptr_t)w3;
    v4u g0 = { 1u,                                   // count=1, no gather
               lds_offset((const void*)sW3f),        // lds_addr
               (unsigned)(ga & 0xFFFFFFFFull),       // global_addr[31:0]
               (unsigned)((ga >> 32) & 0x1FFFFFFull) // global_addr[56:32]
                 | (2u << 30) };                     // type=2 (image)
    v8i g1 = { (int)((2u << 16)        // data_size = 4 bytes
                   | (1u << 20)        // pad_enable
                   | (5u << 22)        // pad_interval: every 64 dwords
                   | (3u << 25)),      // pad_amount: 4 dwords
               (int)0x40000000,        // tensor_dim0 = 16384 (low16 @ [31:16])
               (int)0x00010000,        // dim0 hi=0, tensor_dim1 = 1
               (int)0x40000000,        // dim1 hi=0, tile_dim0 = 16384
               1,                      // tile_dim1 = 1, tile_dim2 = 0
               16384,                  // tensor_dim0_stride
               0, 0 };
    v4i gz4 = { 0, 0, 0, 0 };
    v8i gz8 = { 0, 0, 0, 0, 0, 0, 0, 0 };
    __builtin_amdgcn_tensor_load_to_lds(g0, g1, gz4, gz4, gz8, 0);
    __builtin_amdgcn_s_wait_tensorcnt(0);
  }
#else
  for (int i = tid; i < COUT * CBOX; i += 256)
    sW3f[(i >> 6) * W3PAD + (i & 63)] = w3[i];
#endif

  if (tid < CBOX) {
    float inv = g2[tid] * rsqrtf(v2[tid] + EPSL);
    s_sc2[tid] = inv;
    s_sh2[tid] = bb2[tid] - m2[tid] * inv;
    s_box[tid][0] = ymn_[tid];
    s_box[tid][1] = ymx_[tid];
    s_box[tid][2] = xmn_[tid];
    s_box[tid][3] = xmx_[tid];
  }
  if (tid < COUT) {
    float inv = g3[tid] * rsqrtf(v3[tid] + EPSL);
    s_sc3[tid] = inv;
    s_sh3[tid] = bb3[tid] - m3[tid] * inv;
  }
  __syncthreads();

  int wave = (int)(blockIdx.x * blockDim.x + tid) >> 5;
  int lane = tid & 31;
  int l16  = lane & 15;
  int half = lane >> 4;
  int n  = wave / (HW / 16);
  int p0 = (wave % (HW / 16)) * 16;
  int p  = p0 + l16;
  int hh = p / WW_;
  int wcol = p % WW_;

  // Box features for this lane's pixel, packed directly as B fragments.
  const float* Ibase = I + (size_t)n * CMID * IHW;
  v16h bf[2];
  #pragma unroll
  for (int s = 0; s < 2; ++s) {
    #pragma unroll
    for (int i = 0; i < 16; ++i) {
      int cb = s * 32 + half * 8 + i + (i & 8);     // box channel (K index)
      int cm = cb >> 2;
      const float* Ip = Ibase + (size_t)cm * IHW;
      float ymn = s_box[cb][0], ymx = s_box[cb][1];
      float xmn = s_box[cb][2], xmx = s_box[cb][3];
      float y1 = fminf(fmaxf((float)hh + ymn, 0.f), (float)HH);
      float y2 = fminf(fmaxf((float)hh + ymx + 1.f, 0.f), (float)HH);
      float x1 = fminf(fmaxf((float)wcol + xmn, 0.f), (float)WW_);
      float x2 = fminf(fmaxf((float)wcol + xmx + 1.f, 0.f), (float)WW_);
      float S = corner(Ip, y2, x2) - corner(Ip, y2, x1)
              - corner(Ip, y1, x2) + corner(Ip, y1, x1);
      float area = (ymx - ymn + 1.f) * (xmx - xmn + 1.f);
      float bv = S / area;
      bv = fmaxf(bv * s_sc2[cb] + s_sh2[cb], 0.f);   // BN2 + ReLU
      bf[s][i] = (_Float16)bv;
    }
  }

  const float* xres = x + (size_t)n * COUT * HW + p0 + l16;
  float* op = out + (size_t)n * COUT * HW + p0 + l16;
  for (int mt = 0; mt < 16; ++mt) {                  // 16 Cout tiles
    v8f acc = {};
    #pragma unroll
    for (int s = 0; s < 2; ++s) {                    // K=64 -> 2 WMMA steps
      v16h a;
      const float* wr = sW3f + (mt * 16 + l16) * W3PAD + s * 32 + half * 8;
      #pragma unroll
      for (int i = 0; i < 8; ++i) {
        a[i]     = (_Float16)wr[i];
        a[i + 8] = (_Float16)wr[16 + i];
      }
      acc = __builtin_amdgcn_wmma_f32_16x16x32_f16(false, a, false, bf[s],
                                                   (short)0, acc, false, false);
    }
    #pragma unroll
    for (int r = 0; r < 8; ++r) {
      int co = mt * 16 + r + half * 8;
      float v = acc[r] * s_sc3[co] + s_sh3[co]
              + __builtin_nontemporal_load(xres + (size_t)co * HW);
      __builtin_nontemporal_store(fmaxf(v, 0.f), op + (size_t)co * HW);
    }
  }
}

// ---------------------------------------------------------------------------
extern "C" void kernel_launch(void* const* d_in, const int* in_sizes, int n_in,
                              void* d_out, int out_size, void* d_ws, size_t ws_size,
                              hipStream_t stream) {
  const float* x    = (const float*)d_in[0];
  const float* w1   = (const float*)d_in[1];
  const float* g1   = (const float*)d_in[2];
  const float* b1   = (const float*)d_in[3];
  const float* m1   = (const float*)d_in[4];
  const float* v1   = (const float*)d_in[5];
  const float* ymn  = (const float*)d_in[6];
  const float* ymx  = (const float*)d_in[7];
  const float* xmn  = (const float*)d_in[8];
  const float* xmx  = (const float*)d_in[9];
  const float* g2   = (const float*)d_in[10];
  const float* b2   = (const float*)d_in[11];
  const float* m2   = (const float*)d_in[12];
  const float* v2   = (const float*)d_in[13];
  const float* w3   = (const float*)d_in[14];
  const float* g3   = (const float*)d_in[15];
  const float* b3   = (const float*)d_in[16];
  const float* m3   = (const float*)d_in[17];
  const float* v3   = (const float*)d_in[18];
  float* out = (float*)d_out;

  float* mid = (float*)d_ws;                               // 6.4 MB
  float* I   = mid + (size_t)NB * CMID * HW;               // 6.7 MB

  const int waves1 = NB * (HW / 64);                       // 1568 waves
  k_conv1<<<waves1 / 8, 256, 0, stream>>>(x, w1, g1, b1, m1, v1, mid);
  k_integral<<<NB * CMID, 64, 0, stream>>>(mid, I);
  const int waves3 = NB * (HW / 16);                       // 6272 waves
  k_box_gemm<<<waves3 / 8, 256, 0, stream>>>(I, x, ymn, ymx, xmn, xmx,
                                             g2, b2, m2, v2,
                                             w3, g3, b3, m3, v3, out);
}
